// MinGRU_41815801594305
// MI455X (gfx1250) — matrix operand order repaired and checked by reading, not compiled, not run on hardware
//
#include <hip/hip_runtime.h>
#include <hip/hip_bf16.h>
#include <math.h>

// ---------------------------------------------------------------------------
// MinGRU on MI455X (gfx1250, wave32)
//   B=8, T=4096, D=512, H=512
//   Stage 1: fused GEMM  k = x@Wz^T + bz,  p = x@Wh^T + bh   (bf16x3 WMMA, f32 acc)
//            epilogue -> LC = -softplus(k), LV = -softplus(-k) + log_g(p)
//   Stage 2: 3-phase chunked log-space scan (associative logaddexp recurrence)
// Workspace: LC (64MiB) + LV (64MiB) + chunk summaries (~3MiB) ~= 137 MiB
// ---------------------------------------------------------------------------

#define BB   8
#define TT   4096
#define DD   512
#define HH   512
#define MTOT (BB * TT)        // 32768 rows
#define NC   64               // scan chunks
#define TC   64               // steps per chunk  (NC*TC == TT)

typedef __attribute__((ext_vector_type(16))) __bf16 v16bf;
typedef __attribute__((ext_vector_type(8)))  __bf16 bf16x8;
typedef __attribute__((ext_vector_type(8)))  float  v8f;

union FragU { v16bf v; bf16x8 h[2]; };

__device__ __forceinline__ v8f wmma_bf16(v16bf a, v16bf b, v8f c) {
    // (neg_a, A, neg_b, B, c_mod, C, reuse_a, reuse_b) -> v_wmma_f32_16x16x32_bf16
    return __builtin_amdgcn_wmma_f32_16x16x32_bf16(false, a, false, b, (short)0, c, false, false);
}

__device__ __forceinline__ float softplusf(float v) {
    return fmaxf(v, 0.f) + log1pf(expf(-fabsf(v)));
}
__device__ __forceinline__ float log_gf(float v) {
    return (v >= 0.f) ? logf(v + 0.5f) : -softplusf(-v);
}
__device__ __forceinline__ float logaddexpf_(float a, float b) {
    float mx = fmaxf(a, b);
    if (mx == -INFINITY) return -INFINITY;
    float mn = fminf(a, b);
    return mx + log1pf(expf(mn - mx));
}

// ---------------------------------------------------------------------------
// Stage 1: fused dual-GEMM + epilogue.
// Block tile: 128 (M) x 64 (N over H).  8 waves as 4 (M) x 2 (N); each wave
// owns a 32x32 sub-tile = 2x2 WMMA 16x16 frags, for BOTH the z and h GEMMs.
// bf16x3: x and W split into (hi, lo) bf16; acc += Ahi*Bhi + Ahi*Blo + Alo*Bhi.
// ---------------------------------------------------------------------------
__global__ __launch_bounds__(256)
void mingru_gemm_epilogue(const float* __restrict__ X,
                          const float* __restrict__ Wz,
                          const float* __restrict__ Wh,
                          const float* __restrict__ bz,
                          const float* __restrict__ bh,
                          float* __restrict__ LC,
                          float* __restrict__ LV) {
    __shared__ __bf16 sAhi[128][40];      // 40 = 32 + 8 pad (bank-conflict avoidance)
    __shared__ __bf16 sAlo[128][40];
    __shared__ __bf16 sBhi[2][64][40];    // [0]=Wz tile, [1]=Wh tile
    __shared__ __bf16 sBlo[2][64][40];

    const int tid  = threadIdx.x;
    const int lane = tid & 31;
    const int wave = tid >> 5;
    const int wm   = wave >> 1;           // 0..3
    const int wn   = wave & 1;            // 0..1

    const int m0 = blockIdx.x * 128;      // row block over B*T
    const int n0 = blockIdx.y * 64;       // col block over H

    // global-load mapping: 8 float4 per 32-float row
    const int col4 = tid & 7;             // which float4 in the row
    const int rb   = tid >> 3;            // 0..31

    v8f accz[2][2], acch[2][2];
    #pragma unroll
    for (int i = 0; i < 2; ++i)
        #pragma unroll
        for (int j = 0; j < 2; ++j) { accz[i][j] = (v8f)(0.f); acch[i][j] = (v8f)(0.f); }

    for (int kk = 0; kk < DD; kk += 32) {
        // ---- global loads into registers ----
        float4 aReg[4];
        #pragma unroll
        for (int i = 0; i < 4; ++i) {
            const int row = rb + 32 * i;
            const float* p = X + (size_t)(m0 + row) * DD + kk + col4 * 4;
            aReg[i] = *(const float4*)p;
            if (kk + 32 < DD) __builtin_prefetch(p + 32, 0, 0);   // global_prefetch_b8
        }
        float4 bReg[2][2];
        #pragma unroll
        for (int s = 0; s < 2; ++s) {
            const float* W = (s == 0) ? Wz : Wh;
            #pragma unroll
            for (int i = 0; i < 2; ++i) {
                const int row = rb + 32 * i;  // 0..63
                bReg[s][i] = *(const float4*)(W + (size_t)(n0 + row) * DD + kk + col4 * 4);
            }
        }

        __syncthreads();   // previous iteration's LDS reads done

        // ---- f32 -> bf16 hi/lo split, store to LDS ----
        #pragma unroll
        for (int i = 0; i < 4; ++i) {
            const int row = rb + 32 * i;
            const float* f = (const float*)&aReg[i];
            #pragma unroll
            for (int j = 0; j < 4; ++j) {
                float v = f[j];
                __bf16 hi = (__bf16)v;
                __bf16 lo = (__bf16)(v - (float)hi);
                sAhi[row][col4 * 4 + j] = hi;
                sAlo[row][col4 * 4 + j] = lo;
            }
        }
        #pragma unroll
        for (int s = 0; s < 2; ++s)
            #pragma unroll
            for (int i = 0; i < 2; ++i) {
                const int row = rb + 32 * i;
                const float* f = (const float*)&bReg[s][i];
                #pragma unroll
                for (int j = 0; j < 4; ++j) {
                    float v = f[j];
                    __bf16 hi = (__bf16)v;
                    __bf16 lo = (__bf16)(v - (float)hi);
                    sBhi[s][row][col4 * 4 + j] = hi;
                    sBlo[s][row][col4 * 4 + j] = lo;
                }
            }

        __syncthreads();

        // ---- A fragments (k order used here, same as B: half -> contiguous K chunk h*16..h*16+15) ----
        const int ar   = (lane & 15);
        const int half = lane >> 4;
        FragU ahi[2], alo[2];
        #pragma unroll
        for (int mi = 0; mi < 2; ++mi) {
            const int r = wm * 32 + mi * 16 + ar;
            ahi[mi].h[0] = *(const bf16x8*)&sAhi[r][half * 16];
            ahi[mi].h[1] = *(const bf16x8*)&sAhi[r][half * 16 + 8];
            alo[mi].h[0] = *(const bf16x8*)&sAlo[r][half * 16];
            alo[mi].h[1] = *(const bf16x8*)&sAlo[r][half * 16 + 8];
        }

        // ---- B fragments + WMMA, one weight set at a time (register pressure) ----
        #pragma unroll
        for (int s = 0; s < 2; ++s) {
            FragU bhi[2], blo[2];
            #pragma unroll
            for (int ni = 0; ni < 2; ++ni) {
                const int n  = wn * 32 + ni * 16 + ar;
                const int kc = half * 16;        // B layout: contiguous K halves
                bhi[ni].h[0] = *(const bf16x8*)&sBhi[s][n][kc];
                bhi[ni].h[1] = *(const bf16x8*)&sBhi[s][n][kc + 8];
                blo[ni].h[0] = *(const bf16x8*)&sBlo[s][n][kc];
                blo[ni].h[1] = *(const bf16x8*)&sBlo[s][n][kc + 8];
            }
            #pragma unroll
            for (int mi = 0; mi < 2; ++mi)
                #pragma unroll
                for (int ni = 0; ni < 2; ++ni) {
                    v8f acc = (s == 0) ? accz[mi][ni] : acch[mi][ni];
                    acc = wmma_bf16(ahi[mi].v, bhi[ni].v, acc);
                    acc = wmma_bf16(ahi[mi].v, blo[ni].v, acc);
                    acc = wmma_bf16(alo[mi].v, bhi[ni].v, acc);
                    if (s == 0) accz[mi][ni] = acc; else acch[mi][ni] = acc;
                }
        }
    }

    // ---- fused epilogue: LC = -softplus(k), LV = -softplus(-k) + log_g(p) ----
    const int ar   = (lane & 15);
    const int half = lane >> 4;
    #pragma unroll
    for (int mi = 0; mi < 2; ++mi)
        #pragma unroll
        for (int ni = 0; ni < 2; ++ni) {
            const int h  = n0 + wn * 32 + ni * 16 + ar;
            const float bzv = bz[h];
            const float bhv = bh[h];
            #pragma unroll
            for (int r = 0; r < 8; ++r) {
                const int m = m0 + wm * 32 + mi * 16 + half * 8 + r;
                float kv = accz[mi][ni][r] + bzv;
                float pv = acch[mi][ni][r] + bhv;
                LC[(size_t)m * HH + h] = -softplusf(kv);
                LV[(size_t)m * HH + h] = -softplusf(-kv) + log_gf(pv);
            }
        }
}

// ---------------------------------------------------------------------------
// Stage 2a: per-chunk scan summaries.  Segment op over chunk c of channel (b,h):
//   out = logaddexp(A + in, V),  A = sum lc,  V = running logaddexp.
// One thread per (b, c, h); consecutive h -> coalesced.
// ---------------------------------------------------------------------------
__global__ __launch_bounds__(256)
void mingru_scan_chunks(const float* __restrict__ LC, const float* __restrict__ LV,
                        float* __restrict__ CA, float* __restrict__ CV) {
    const int g   = blockIdx.x * blockDim.x + threadIdx.x;
    const int h   = g % HH;
    const int tmp = g / HH;
    const int c   = tmp % NC;
    const int b   = tmp / NC;

    size_t idx = ((size_t)b * TT + (size_t)c * TC) * HH + h;
    float A = 0.f, V = -INFINITY;
    #pragma unroll 4
    for (int t = 0; t < TC; ++t) {
        float lc = LC[idx];
        float lv = LV[idx];
        A += lc;
        V = logaddexpf_(lc + V, lv);
        idx += HH;
    }
    const size_t o = ((size_t)b * NC + c) * HH + h;
    CA[o] = A;
    CV[o] = V;
}

// ---------------------------------------------------------------------------
// Stage 2b: sequential combine over NC chunk summaries per channel;
// records each chunk's incoming state.  4096 threads, 64 cheap steps.
// ---------------------------------------------------------------------------
__global__ __launch_bounds__(256)
void mingru_scan_combine(const float* __restrict__ h0,
                         const float* __restrict__ CA, const float* __restrict__ CV,
                         float* __restrict__ SIN) {
    const int g = blockIdx.x * blockDim.x + threadIdx.x;   // b*H + h
    const int h = g % HH;
    const int b = g / HH;

    float s = log_gf(h0[(size_t)b * HH + h]);
    for (int c = 0; c < NC; ++c) {
        const size_t o = ((size_t)b * NC + c) * HH + h;
        SIN[o] = s;
        s = logaddexpf_(CA[o] + s, CV[o]);
    }
}

// ---------------------------------------------------------------------------
// Stage 2c: re-run each chunk with the correct incoming state, emit exp(log_h).
// ---------------------------------------------------------------------------
__global__ __launch_bounds__(256)
void mingru_scan_emit(const float* __restrict__ LC, const float* __restrict__ LV,
                      const float* __restrict__ SIN, float* __restrict__ out) {
    const int g   = blockIdx.x * blockDim.x + threadIdx.x;
    const int h   = g % HH;
    const int tmp = g / HH;
    const int c   = tmp % NC;
    const int b   = tmp / NC;

    float lh = SIN[((size_t)b * NC + c) * HH + h];
    size_t idx = ((size_t)b * TT + (size_t)c * TC) * HH + h;
    #pragma unroll 4
    for (int t = 0; t < TC; ++t) {
        lh = logaddexpf_(LC[idx] + lh, LV[idx]);
        out[idx] = expf(lh);
        idx += HH;
    }
}

// ---------------------------------------------------------------------------
extern "C" void kernel_launch(void* const* d_in, const int* in_sizes, int n_in,
                              void* d_out, int out_size, void* d_ws, size_t ws_size,
                              hipStream_t stream) {
    (void)in_sizes; (void)n_in; (void)out_size; (void)ws_size;

    const float* x  = (const float*)d_in[0];
    const float* h0 = (const float*)d_in[1];
    const float* Wz = (const float*)d_in[2];
    const float* bz = (const float*)d_in[3];
    const float* Wh = (const float*)d_in[4];
    const float* bh = (const float*)d_in[5];
    float* out = (float*)d_out;

    // workspace layout (floats): LC | LV | CA | CV | SIN   (~137 MiB total)
    float* ws  = (float*)d_ws;
    const size_t NEL = (size_t)MTOT * HH;        // 16,777,216
    const size_t NCH = (size_t)BB * HH * NC;     // 262,144
    float* LC  = ws;
    float* LV  = LC + NEL;
    float* CA  = LV + NEL;
    float* CV  = CA + NCH;
    float* SIN = CV + NCH;

    dim3 g1(MTOT / 128, HH / 64);                // 256 x 8 blocks
    mingru_gemm_epilogue<<<g1, 256, 0, stream>>>(x, Wz, Wh, bz, bh, LC, LV);

    mingru_scan_chunks <<<(int)(NCH / 256), 256, 0, stream>>>(LC, LV, CA, CV);
    mingru_scan_combine<<<(BB * HH) / 256,   256, 0, stream>>>(h0, CA, CV, SIN);
    mingru_scan_emit   <<<(int)(NCH / 256), 256, 0, stream>>>(LC, LV, SIN, out);
}
